// CM_29540785062535
// MI455X (gfx1250) — matrix operand order repaired and checked by reading, not compiled, not run on hardware
//
#include <hip/hip_runtime.h>
#include <hip/hip_bf16.h>

// ---------------------------------------------------------------------------
// Fused gated-context block for MI455X (gfx1250, wave32, WMMA).
//   fx   = f_w @ x + f_b            (GEMM1, bf16 WMMA, fp32 accum)
//   ctx  = sum_l x_list[l]*gates[l] (elementwise, streams 604MB once, b128)
//   mod  = h_w @ ctx + h_b          (GEMM2, mt-outer; q folded in registers)
//   y    = proj_w @ (q*mod) + p_b   (GEMM3)
//   out  = LN_c(y)*ln_w + ln_b + x  (register LN via shfl_xor(16), NT stores)
// Active weight matrix staged in LDS (once per WG, overlapped with other
// phases); A fragments are ds_load_b128; q / x_out / y live in registers.
// ---------------------------------------------------------------------------

typedef __attribute__((ext_vector_type(16))) __bf16 v16bf;
typedef __attribute__((ext_vector_type(8)))  __bf16 v8bf;
typedef __attribute__((ext_vector_type(8)))  float  v8f;
typedef __attribute__((ext_vector_type(4)))  float  v4f;

#define C_CH    192
#define NIMG    16
#define HW      16384          // 128*128
#define PIXT    64             // pixels per workgroup
#define PITCH_F 194            // fp32 LDS row pitch (floats)
#define PITCH_H 200            // bf16 LDS row pitch (elements)
#define W_PITCH 200            // bf16 LDS weight pitch (rows start on 16 distinct banks)
#define NTHREADS 128

// bf16 weight workspace layout (elements): f_w padded to 208 rows, then h_w, proj_w
#define WSF_ROWS 208
#define WS_F_OFF 0
#define WS_H_OFF (WSF_ROWS * C_CH)              // 39936
#define WS_P_OFF (WS_H_OFF + C_CH * C_CH)       // 76800
#define WS_ELEMS (WS_P_OFF + C_CH * C_CH)       // 113664 (227,328 bytes of d_ws)

// LDS carve-out (dynamic shared memory) -- total 163,392 B -> 2 WGs / 320KB WGP
#define OFF_SW 0                                   // bf16 staged weights [208][200]
#define OFF_SX (OFF_SW + WSF_ROWS * W_PITCH * 2)   // fp32 x tile   [64][194]   (83,200)
#define OFF_SB (OFF_SX + PIXT * PITCH_F * 4)       // bf16 ctx/x_out [64][200]  (132,864)
#define OFF_SG (OFF_SB + PIXT * PITCH_H * 2)       // fp32 gates [64][4]        (158,464)
#define OFF_SP (OFF_SG + PIXT * 4 * 4)             // fp32 params               (159,488)
#define PB_F   0                                   // f_b   (195, padded 208)
#define PB_H   208                                 // h_b   (192)
#define PB_P   400                                 // p_b   (192)
#define PB_LW  592                                 // ln_w  (192)
#define PB_LB  784                                 // ln_b  (192)
#define NPARAM 976
#define SMEM_BYTES (OFF_SP + NPARAM * 4)           // 163,392

// ---- prep: fp32 weights -> packed bf16 in workspace (f_w zero-padded to 208 rows)
__global__ void CM_prep_weights(const float* __restrict__ f_w,
                                const float* __restrict__ h_w,
                                const float* __restrict__ p_w,
                                __bf16* __restrict__ ws) {
  int i = blockIdx.x * 256 + threadIdx.x;
  if (i < WS_H_OFF) {                      // padded f_w block
    int r = i / C_CH;
    ws[i] = (__bf16)((r < 195) ? f_w[i] : 0.0f);
  } else if (i < WS_P_OFF) {
    ws[i] = (__bf16)h_w[i - WS_H_OFF];
  } else if (i < WS_ELEMS) {
    ws[i] = (__bf16)p_w[i - WS_P_OFF];
  }
}

// Copy a [rows][192] bf16 weight matrix from global into LDS (pitch W_PITCH).
__device__ __forceinline__ void stage_weights(const __bf16* __restrict__ src, int rows,
                                              __bf16* __restrict__ dst, int tid) {
  for (int i = tid; i < rows * 24; i += NTHREADS) {   // 24 x v8bf chunks per row
    int r = i / 24, ch = (i % 24) * 8;
    *(v8bf*)(dst + r * W_PITCH + ch) = *(const v8bf*)(src + r * C_CH + ch);
  }
}

// A fragment: 16x32 (MxK) from LDS-staged bf16 weights.
// lane&15 = M row; per-lane data = two contiguous 16B runs -> 2x ds_load_b128.
__device__ __forceinline__ v16bf afrag_lds(const __bf16* __restrict__ sW,
                                           int mBase, int kBase, int lane) {
  const __bf16* wr = sW + (mBase + (lane & 15)) * W_PITCH + kBase + (((lane >> 4) & 1) * 16);
  v8bf lo = *(const v8bf*)(wr);
  v8bf hi = *(const v8bf*)(wr + 8);
  return __builtin_shufflevector(lo, hi, 0,1,2,3,4,5,6,7,8,9,10,11,12,13,14,15);
}

// B fragment: 32x16 (KxN). lane&15 = column pixel; lanes>=16 cover K+16..K+31.
__device__ __forceinline__ v16bf bfrag_f32(const float* __restrict__ row,
                                           int kBase, int lane) {
  const float* p = row + kBase + (((lane >> 4) & 1) * 16);
  v16bf b;
#pragma unroll
  for (int e = 0; e < 16; ++e) b[e] = (__bf16)p[e];
  return b;
}

__device__ __forceinline__ v16bf bfrag_h(const __bf16* __restrict__ row,
                                         int kBase, int lane) {
  const __bf16* p = row + kBase + (((lane >> 4) & 1) * 16);   // 16B aligned
  v8bf lo = *(const v8bf*)(p);
  v8bf hi = *(const v8bf*)(p + 8);
  return __builtin_shufflevector(lo, hi, 0,1,2,3,4,5,6,7,8,9,10,11,12,13,14,15);
}

__global__ __launch_bounds__(NTHREADS, 1)
void CM_fused_gated_block(const float* __restrict__ x,
                          const float* __restrict__ x_list,
                          const __bf16* __restrict__ wb,   // bf16 weights in d_ws
                          const float* __restrict__ f_b,
                          const float* __restrict__ h_b,
                          const float* __restrict__ p_b,
                          const float* __restrict__ ln_w, const float* __restrict__ ln_b,
                          float* __restrict__ out) {
  extern __shared__ unsigned char smem[];
  __bf16* sW = (__bf16*)(smem + OFF_SW);
  float*  sX = (float*)(smem + OFF_SX);
  __bf16* sB = (__bf16*)(smem + OFF_SB);
  float*  sG = (float*)(smem + OFF_SG);
  float*  sP = (float*)(smem + OFF_SP);

  const int tid  = threadIdx.x;
  const int lane = tid & 31;
  const int wave = tid >> 5;
  const int pixBase = blockIdx.x * PIXT;
  const int n   = pixBase >> 14;       // tile fully inside one image (HW % 64 == 0)
  const int hw0 = pixBase & (HW - 1);

  // ---- Phase A: x tile -> sX (b128), f_w -> sW, params -> sP ----
  for (int i = tid; i < PIXT * C_CH / 4; i += NTHREADS) {
    int p4 = (i & 15) * 4;
    int c  = i >> 4;
    const v4f* src = (const v4f*)(x + (((size_t)(n * C_CH + c)) << 14) + hw0 + p4);
    v4f v = __builtin_nontemporal_load(src);
#pragma unroll
    for (int j = 0; j < 4; ++j) sX[(p4 + j) * PITCH_F + c] = v[j];
  }
  stage_weights(wb + WS_F_OFF, WSF_ROWS, sW, tid);
  for (int i = tid; i < 195; i += NTHREADS) sP[PB_F + i] = f_b[i];
  for (int i = tid; i < C_CH; i += NTHREADS) {
    sP[PB_H + i]  = h_b[i];
    sP[PB_P + i]  = p_b[i];
    sP[PB_LW + i] = ln_w[i];
    sP[PB_LB + i] = ln_b[i];
  }
  __syncthreads();

  const int pl   = lane & 15;
  const int p    = wave * 16 + pl;          // this wave's B-operand column (pixel)
  const int mOff = ((lane >> 4) & 1) * 8;   // C/D row offset for upper half-wave

  // ---- Phase B: GEMM1  fx = f_w @ x  (kt-outer, 13 accs; q stays in registers) ----
  v8f fxa[13] = {};
#pragma unroll
  for (int kt = 0; kt < 6; ++kt) {
    v16bf b = bfrag_f32(sX + p * PITCH_F, kt * 32, lane);
#pragma unroll
    for (int mt = 0; mt < 13; ++mt) {
      v16bf a = afrag_lds(sW, mt * 16, kt * 32, lane);
      fxa[mt] = __builtin_amdgcn_wmma_f32_16x16x32_bf16(
          false, a, false, b, (short)0, fxa[mt], false, false);
    }
  }
  if (mOff == 0) {                          // rows 192..194 of fx are the L=3 gates
#pragma unroll
    for (int r = 0; r < 3; ++r) sG[p * 4 + r] = fxa[12][r] + sP[PB_F + C_CH + r];
  }
  __syncthreads();

  // ---- Phase C: ctx -> sB (b128 x_list streams); h_w -> sW (overlapped) ----
  {
    const size_t lstride = (size_t)NIMG * C_CH * HW;     // one x_list layer
#pragma unroll 2
    for (int i = tid; i < (PIXT / 4) * C_CH; i += NTHREADS) {
      int p4 = (i & 15) * 4;
      int c  = i >> 4;
      size_t base = (((size_t)(n * C_CH + c)) << 14) + hw0 + p4;
      v4f a0 = __builtin_nontemporal_load((const v4f*)(x_list + base));
      v4f a1 = __builtin_nontemporal_load((const v4f*)(x_list + base + lstride));
      v4f a2 = __builtin_nontemporal_load((const v4f*)(x_list + base + 2 * lstride));
#pragma unroll
      for (int j = 0; j < 4; ++j) {
        int pp = p4 + j;
        float acc = a0[j] * sG[pp * 4 + 0] + a1[j] * sG[pp * 4 + 1]
                  + a2[j] * sG[pp * 4 + 2];
        sB[pp * PITCH_H + c] = (__bf16)acc;
      }
    }
    stage_weights(wb + WS_H_OFF, C_CH, sW, tid);
  }
  __syncthreads();

  // ---- Phase D: GEMM2 (mt-outer, one macc live) + fold x_out = (q+f_b)*(mod+h_b) ----
#pragma unroll
  for (int mt = 0; mt < 12; ++mt) {
    v8f macc = {};
#pragma unroll
    for (int kt = 0; kt < 6; ++kt) {
      v16bf b = bfrag_h(sB + p * PITCH_H, kt * 32, lane);
      v16bf a = afrag_lds(sW, mt * 16, kt * 32, lane);
      macc = __builtin_amdgcn_wmma_f32_16x16x32_bf16(
          false, a, false, b, (short)0, macc, false, false);
    }
#pragma unroll
    for (int r = 0; r < 8; ++r) {
      int m = mt * 16 + mOff + r;
      float q = fxa[mt][r] + sP[PB_F + m];
      fxa[mt][r] = q * (macc[r] + sP[PB_H + m]);   // fxa now holds x_out
    }
  }
  __syncthreads();                                 // all waves done reading ctx in sB

  // ---- Phase E: x_out -> sB (bf16, reuse); proj_w -> sW (overlapped) ----
#pragma unroll
  for (int mt = 0; mt < 12; ++mt)
#pragma unroll
    for (int r = 0; r < 8; ++r)
      sB[p * PITCH_H + mt * 16 + mOff + r] = (__bf16)fxa[mt][r];
  stage_weights(wb + WS_P_OFF, C_CH, sW, tid);
  __syncthreads();

  // ---- Phase F: GEMM3  y = proj_w @ x_out + p_b  (kt-outer, y in registers) ----
  v8f y[12] = {};
#pragma unroll
  for (int kt = 0; kt < 6; ++kt) {
    v16bf b = bfrag_h(sB + p * PITCH_H, kt * 32, lane);
#pragma unroll
    for (int mt = 0; mt < 12; ++mt) {
      v16bf a = afrag_lds(sW, mt * 16, kt * 32, lane);
      y[mt] = __builtin_amdgcn_wmma_f32_16x16x32_bf16(
          false, a, false, b, (short)0, y[mt], false, false);
    }
  }

  // ---- Phase G: register LayerNorm (lanes pl / pl+16 hold pixel p's 2x96 chans) ----
  float s = 0.0f;
#pragma unroll
  for (int mt = 0; mt < 12; ++mt)
#pragma unroll
    for (int r = 0; r < 8; ++r) {
      y[mt][r] += sP[PB_P + mt * 16 + mOff + r];
      s += y[mt][r];
    }
  float mean = (s + __shfl_xor(s, 16, 32)) * (1.0f / C_CH);
  float ss = 0.0f;
#pragma unroll
  for (int mt = 0; mt < 12; ++mt)
#pragma unroll
    for (int r = 0; r < 8; ++r) {
      float d = y[mt][r] - mean;
      ss += d * d;
    }
  float rstd = rsqrtf((ss + __shfl_xor(ss, 16, 32)) * (1.0f / C_CH) + 1e-6f);

  const size_t obase = (((size_t)n * C_CH) << 14) + hw0 + p;
#pragma unroll
  for (int mt = 0; mt < 12; ++mt)
#pragma unroll
    for (int r = 0; r < 8; ++r) {
      int m = mt * 16 + mOff + r;
      float v = (y[mt][r] - mean) * rstd * sP[PB_LW + m] + sP[PB_LB + m]
              + sX[p * PITCH_F + m];
      __builtin_nontemporal_store(v, out + obase + ((size_t)m << 14));
    }
}

extern "C" void kernel_launch(void* const* d_in, const int* in_sizes, int n_in,
                              void* d_out, int out_size, void* d_ws, size_t ws_size,
                              hipStream_t stream) {
  (void)n_in; (void)ws_size; (void)out_size;
  const float* x      = (const float*)d_in[0];
  const float* x_list = (const float*)d_in[1];
  const float* f_w    = (const float*)d_in[2];
  const float* f_b    = (const float*)d_in[3];
  const float* h_w    = (const float*)d_in[4];
  const float* h_b    = (const float*)d_in[5];
  const float* p_w    = (const float*)d_in[6];
  const float* p_b    = (const float*)d_in[7];
  const float* ln_w   = (const float*)d_in[8];
  const float* ln_b   = (const float*)d_in[9];
  float*  out = (float*)d_out;
  __bf16* wb  = (__bf16*)d_ws;                   // needs 227,328 bytes of scratch

  // 1) fp32 -> bf16 weight conversion (tiny; L2-resident afterwards)
  CM_prep_weights<<<(WS_ELEMS + 255) / 256, 256, 0, stream>>>(f_w, h_w, p_w, wb);

  // 2) fused block
  const int total_pixels = in_sizes[0] / C_CH;   // N*H*W = 262144
  const int blocks = total_pixels / PIXT;        // 4096

  (void)hipFuncSetAttribute((const void*)CM_fused_gated_block,
                            hipFuncAttributeMaxDynamicSharedMemorySize, SMEM_BYTES);

  CM_fused_gated_block<<<blocks, NTHREADS, SMEM_BYTES, stream>>>(
      x, x_list, wb, f_b, h_b, p_b, ln_w, ln_b, out);
}
